// APPNPNet_88811333746741
// MI455X (gfx1250) — matrix operand order repaired and checked by reading, not compiled, not run on hardware
//
#include <hip/hip_runtime.h>
#include <hip/hip_bf16.h>

typedef __bf16 bf16_t;
typedef __attribute__((ext_vector_type(16))) __bf16 v16bf;
typedef __attribute__((ext_vector_type(8)))  float  v8f;
typedef __attribute__((ext_vector_type(8)))  unsigned int v8u;

#define IN_C 512
#define HID  256
#define OUTC 64
#define LDSROW 257   // u32 stride per LDS row (256 data + 1 pad -> conflict free)

__device__ __forceinline__ unsigned int f2bf(float f) {
    unsigned int u = __builtin_bit_cast(unsigned int, f);
    unsigned int r = u + 0x7FFFu + ((u >> 16) & 1u);   // round-to-nearest-even
    return r >> 16;
}

__device__ __forceinline__ void atomic_add_f32(float* p, float v) {
    // hardware FP32 atomic (no return) — tracked with STOREcnt
    asm volatile("global_atomic_add_f32 %0, %1, off" :: "v"(p), "v"(v) : "memory");
}

// ---------------------------------------------------------------------------
// Pre-pack fp32 weight matrix [K x ncols] into WMMA bf16 B-fragment order.
// out[((kblk*nct + ct)*32 + lane)*8 + v] = pack(W[K0,n], W[K0+1,n])
//   K0 = kblk*32 + (lane>>4)*8 + 2(v&3) + 16(v>>2),  n = ct*16 + (lane&15)
// ---------------------------------------------------------------------------
__global__ void prepack_kernel(const float* __restrict__ W, unsigned int* __restrict__ out,
                               int ncols, int nct, int total) {
    int tid = blockIdx.x * blockDim.x + threadIdx.x;
    if (tid >= total) return;
    int v    = tid & 7;
    int L    = (tid >> 3) & 31;
    int ct   = (tid >> 8) % nct;
    int kblk = tid / (256 * nct);
    int K0 = kblk * 32 + (L >> 4) * 8 + 2 * (v & 3) + 16 * (v >> 2);
    int n  = ct * 16 + (L & 15);
    unsigned lo = f2bf(W[(size_t)K0 * ncols + n]);
    unsigned hi = f2bf(W[(size_t)(K0 + 1) * ncols + n]);
    out[tid] = lo | (hi << 16);
}

// ---------------------------------------------------------------------------
// GEMM1: h1 = relu(x @ W1 + b1), bf16 output [N, 256]
// block = 16 rows x 256 cols; 8 waves, 2 col-tiles each; x staged in LDS as bf16
// ---------------------------------------------------------------------------
__global__ __launch_bounds__(256) void gemm1_kernel(
    const float* __restrict__ x, const unsigned int* __restrict__ w1p,
    const float* __restrict__ bias1, unsigned short* __restrict__ h1, int N)
{
    __shared__ unsigned int lds[16 * LDSROW];
    int row0 = blockIdx.x * 16;
    if (row0 >= N) return;
    int tid = threadIdx.x;

    // stage 16 x 512 fp32 -> bf16 pairs in LDS (16 u32 per thread)
    for (int i = tid; i < 16 * 256; i += 256) {
        int r = i >> 8;
        int cp = i & 255;
        int rr = row0 + r; if (rr >= N) rr = N - 1;
        const float* px = x + (size_t)rr * IN_C + cp * 2;
        lds[r * LDSROW + cp] = f2bf(px[0]) | (f2bf(px[1]) << 16);
    }
    __syncthreads();

    int wave = tid >> 5, lane = tid & 31;
    int m = lane & 15, g = lane >> 4;
    int ct0 = wave * 2;

    v8f c0 = {}; v8f c1 = {};
    for (int kk = 0; kk < IN_C / 32; ++kk) {
        v8u au;
        int abase = m * LDSROW + kk * 16 + 4 * g;
        #pragma unroll
        for (int v = 0; v < 8; ++v)
            au[v] = lds[abase + (v & 3) + 8 * (v >> 2)];
        v16bf a = __builtin_bit_cast(v16bf, au);
        v16bf b0 = __builtin_bit_cast(v16bf, *(const v8u*)(w1p + ((kk * 16 + ct0)     * 32 + lane) * 8));
        v16bf b1 = __builtin_bit_cast(v16bf, *(const v8u*)(w1p + ((kk * 16 + ct0 + 1) * 32 + lane) * 8));
        c0 = __builtin_amdgcn_wmma_f32_16x16x32_bf16(false, a, false, b0, (short)0, c0, false, false);
        c1 = __builtin_amdgcn_wmma_f32_16x16x32_bf16(false, a, false, b1, (short)0, c1, false, false);
    }

    int n = lane & 15, mg = (lane >> 4) * 8;
    float bia0 = bias1[ct0 * 16 + n];
    float bia1 = bias1[(ct0 + 1) * 16 + n];
    #pragma unroll
    for (int r = 0; r < 8; ++r) {
        int row = row0 + mg + r;
        if (row >= N) continue;
        float v0 = c0[r] + bia0; v0 = v0 > 0.0f ? v0 : 0.0f;
        float v1 = c1[r] + bia1; v1 = v1 > 0.0f ? v1 : 0.0f;
        h1[(size_t)row * HID + ct0 * 16 + n]       = (unsigned short)f2bf(v0);
        h1[(size_t)row * HID + (ct0 + 1) * 16 + n] = (unsigned short)f2bf(v1);
    }
}

// ---------------------------------------------------------------------------
// GEMM2: h0 = h1 @ W2 + b2, fp32 output [N, 64]
// block = 32 rows x 64 cols; 8 waves = 2 row-tiles x 4 col-tiles
// ---------------------------------------------------------------------------
__global__ __launch_bounds__(256) void gemm2_kernel(
    const unsigned int* __restrict__ h1u, const unsigned int* __restrict__ w2p,
    const float* __restrict__ bias2, float* __restrict__ h0, int N)
{
    int tid = threadIdx.x;
    int wave = tid >> 5, lane = tid & 31;
    int rt = wave >> 2, ct = wave & 3;
    int row0 = blockIdx.x * 32 + rt * 16;
    if (row0 >= N) return;
    int m = lane & 15, g = lane >> 4;
    int rr = row0 + m; if (rr >= N) rr = N - 1;

    v8f c = {};
    for (int kk = 0; kk < HID / 32; ++kk) {
        v8u au;
        const unsigned int* ap = h1u + (size_t)rr * (HID / 2) + kk * 16 + 4 * g;
        #pragma unroll
        for (int v = 0; v < 8; ++v)
            au[v] = ap[(v & 3) + 8 * (v >> 2)];
        v16bf a = __builtin_bit_cast(v16bf, au);
        v16bf b = __builtin_bit_cast(v16bf, *(const v8u*)(w2p + ((kk * 4 + ct) * 32 + lane) * 8));
        c = __builtin_amdgcn_wmma_f32_16x16x32_bf16(false, a, false, b, (short)0, c, false, false);
    }

    int n = lane & 15, mg = (lane >> 4) * 8;
    float bia = bias2[ct * 16 + n];
    #pragma unroll
    for (int r = 0; r < 8; ++r) {
        int row = row0 + mg + r;
        if (row < N) h0[(size_t)row * OUTC + ct * 16 + n] = c[r] + bia;
    }
}

// ---------------------------------------------------------------------------
// gcn_norm helpers
// ---------------------------------------------------------------------------
__global__ void deg_init_kernel(float* deg, int N) {
    int i = blockIdx.x * blockDim.x + threadIdx.x;
    if (i < N) deg[i] = 1.0f;                     // self loop
}
__global__ void deg_acc_kernel(const int* __restrict__ dst, float* deg, int E) {
    int e = blockIdx.x * blockDim.x + threadIdx.x;
    if (e < E) atomic_add_f32(&deg[dst[e]], 1.0f);
}
__global__ void dinv_kernel(const float* __restrict__ deg, float* dinv, int N) {
    int i = blockIdx.x * blockDim.x + threadIdx.x;
    if (i < N) dinv[i] = rsqrtf(deg[i]);
}
// copy i32 indices + precompute per-edge norm (once; reused 10x)
__global__ void edge_pre_kernel(const int* __restrict__ src, const int* __restrict__ dst,
                                const float* __restrict__ dinv,
                                int* srcI, int* dstI, float* normE, int E) {
    int e = blockIdx.x * blockDim.x + threadIdx.x;
    if (e >= E) return;
    int s = src[e], d = dst[e];
    srcI[e] = s; dstI[e] = d;
    normE[e] = dinv[s] * dinv[d];
}

// out = ALPHA*h0 + (1-ALPHA)*dinv^2*hcur   (self-loop term folded in, no atomics)
__global__ void iter_init_kernel(const float* __restrict__ h0, const float* __restrict__ hcur,
                                 const float* __restrict__ dinv, float* __restrict__ out, int total) {
    int i = blockIdx.x * blockDim.x + threadIdx.x;
    if (i >= total) return;
    float di = dinv[i >> 6];
    out[i] = 0.1f * h0[i] + 0.9f * di * di * hcur[i];
}

// 16 lanes per edge, float4 each: out[dst] += 0.9*norm*h[src]
__global__ __launch_bounds__(256) void edge_prop_kernel(
    const int* __restrict__ srcI, const int* __restrict__ dstI, const float* __restrict__ normE,
    const float* __restrict__ h, float* __restrict__ out, int E)
{
    int t = blockIdx.x * 256 + threadIdx.x;
    int e = t >> 4;
    if (e >= E) return;
    int f = (t & 15) * 4;
    int s = srcI[e], d = dstI[e];
    float w = 0.9f * normE[e];
    float4 hv = *(const float4*)(h + (size_t)s * OUTC + f);
    float* o = out + (size_t)d * OUTC + f;
    atomic_add_f32(o + 0, w * hv.x);
    atomic_add_f32(o + 1, w * hv.y);
    atomic_add_f32(o + 2, w * hv.z);
    atomic_add_f32(o + 3, w * hv.w);
}

// ---------------------------------------------------------------------------
extern "C" void kernel_launch(void* const* d_in, const int* in_sizes, int n_in,
                              void* d_out, int out_size, void* d_ws, size_t ws_size,
                              hipStream_t stream)
{
    const float* x   = (const float*)d_in[0];
    const int*   ei  = (const int*)d_in[1];    // (2,E) int32
    const float* W1  = (const float*)d_in[2];
    const float* bb1 = (const float*)d_in[3];
    const float* W2  = (const float*)d_in[4];
    const float* bb2 = (const float*)d_in[5];

    const int N = in_sizes[0] / IN_C;
    const int E = in_sizes[1] / 2;
    const int* src = ei;
    const int* dst = ei + E;

    char* ws = (char*)d_ws;
    size_t off = 0;
    auto alloc = [&](size_t bytes) -> void* {
        void* p = ws + off;
        off = (off + bytes + 255) & ~(size_t)255;
        return p;
    };
    float* h0    = (float*)alloc((size_t)N * OUTC * 4);
    float* hA    = (float*)alloc((size_t)N * OUTC * 4);
    float* hB    = (float*)alloc((size_t)N * OUTC * 4);
    float* deg   = (float*)alloc((size_t)N * 4);
    float* dinv  = (float*)alloc((size_t)N * 4);
    int*   srcI  = (int*)alloc((size_t)E * 4);
    int*   dstI  = (int*)alloc((size_t)E * 4);
    float* normE = (float*)alloc((size_t)E * 4);
    unsigned short* h1 = (unsigned short*)alloc((size_t)N * HID * 2);
    const int W1P = (IN_C / 32) * (HID / 16) * 32 * 8;   // 65536 u32
    const int W2P = (HID / 32) * (OUTC / 16) * 32 * 8;   // 8192 u32
    unsigned int* w1p = (unsigned int*)alloc((size_t)W1P * 4);
    unsigned int* w2p = (unsigned int*)alloc((size_t)W2P * 4);

    // weight pre-pack into WMMA B-fragment order (bf16)
    prepack_kernel<<<(W1P + 255) / 256, 256, 0, stream>>>(W1, w1p, HID,  HID / 16,  W1P);
    prepack_kernel<<<(W2P + 255) / 256, 256, 0, stream>>>(W2, w2p, OUTC, OUTC / 16, W2P);

    // MLP encoder: two WMMA GEMMs
    gemm1_kernel<<<(N + 15) / 16, 256, 0, stream>>>(x, w1p, bb1, h1, N);
    gemm2_kernel<<<(N + 31) / 32, 256, 0, stream>>>((const unsigned int*)h1, w2p, bb2, h0, N);

    // gcn_norm
    deg_init_kernel<<<(N + 255) / 256, 256, 0, stream>>>(deg, N);
    deg_acc_kernel <<<(E + 255) / 256, 256, 0, stream>>>(dst, deg, E);
    dinv_kernel    <<<(N + 255) / 256, 256, 0, stream>>>(deg, dinv, N);
    edge_pre_kernel<<<(E + 255) / 256, 256, 0, stream>>>(src, dst, dinv, srcI, dstI, normE, E);

    // K = 10 propagation iterations (ping-pong; last one writes d_out)
    const int total = N * OUTC;
    const int prop_blocks = (int)(((size_t)E * 16 + 255) / 256);
    const float* cur = h0;
    for (int k = 0; k < 10; ++k) {
        float* out = (k == 9) ? (float*)d_out : ((k & 1) ? hB : hA);
        iter_init_kernel<<<(total + 255) / 256, 256, 0, stream>>>(h0, cur, dinv, out, total);
        edge_prop_kernel<<<prop_blocks, 256, 0, stream>>>(srcI, dstI, normE, cur, out, E);
        cur = out;
    }
}
